// BoundaryGraphPredictor_2104533975798
// MI455X (gfx1250) — matrix operand order repaired and checked by reading, not compiled, not run on hardware
//
#include <hip/hip_runtime.h>
#include <hip/hip_bf16.h>

typedef __attribute__((ext_vector_type(16))) __bf16 v16bf;
typedef __attribute__((ext_vector_type(8)))  float  v8f;

#define NN   20000
#define DIMM 512
#define HH   8
#define EE   320000
#define COLS 2048            // 4 * 512 fused output columns
#define KT   16              // 512 / 32 k-tiles
#define NT   128             // 2048 / 16 n-tiles

// workspace layout (float units)
static const size_t OFF_Q   = 0;
static const size_t OFF_K   = OFF_Q  + (size_t)NN * DIMM;      // 10,240,000
static const size_t OFF_V   = OFF_K  + (size_t)NN * DIMM;
static const size_t OFF_S   = OFF_V  + (size_t)NN * DIMM;      // skip
static const size_t OFF_AGG = OFF_S  + (size_t)NN * DIMM;
static const size_t OFF_SC  = OFF_AGG + (size_t)NN * DIMM;     // scores / ex  (E*H)
static const size_t OFF_MAX = OFF_SC + (size_t)EE * HH;        // seg max (encoded uint, N*H)
static const size_t OFF_DEN = OFF_MAX + (size_t)NN * HH;       // denom (N*H)
static const size_t OFF_WP  = OFF_DEN + (size_t)NN * HH;       // packed bf16 weights (2MB)

__device__ __forceinline__ unsigned encOrd(float f) {
    unsigned u = __float_as_uint(f);
    return (u & 0x80000000u) ? ~u : (u | 0x80000000u);
}
__device__ __forceinline__ float decOrd(unsigned e) {
    unsigned u = (e & 0x80000000u) ? (e ^ 0x80000000u) : ~e;
    return __uint_as_float(u);
}
__device__ __forceinline__ void atomAddF(float* p, float v) {
    __hip_atomic_fetch_add(p, v, __ATOMIC_RELAXED, __HIP_MEMORY_SCOPE_AGENT);
}

// ---- pack four [512,512] f32 weight mats into WMMA B-fragment bf16 layout ----
__global__ void bgp_pack_w(const float* __restrict__ Wq, const float* __restrict__ Wk,
                           const float* __restrict__ Wv, const float* __restrict__ Ws,
                           __bf16* __restrict__ Wp) {
    int idx = blockIdx.x * blockDim.x + threadIdx.x;      // kt*128*32 + ntile*32 + lane
    if (idx >= KT * NT * 32) return;
    int lane  = idx & 31;
    int ntile = (idx >> 5) & (NT - 1);
    int kt    = idx >> 12;
    int col = ntile * 16 + (lane & 15);
    int seg = col >> 9, cid = col & 511;
    const float* W = (seg == 0) ? Wq : (seg == 1) ? Wk : (seg == 2) ? Wv : Ws;
    int klo = kt * 32 + ((lane & 16) ? 16 : 0);
    __bf16* o = Wp + (size_t)idx * 16;
    #pragma unroll
    for (int j = 0; j < 16; ++j) o[j] = (__bf16)W[(size_t)(klo + j) * 512 + cid];
}

__global__ void bgp_fill(float* p, float val, size_t n) {
    size_t i = (size_t)blockIdx.x * blockDim.x + threadIdx.x;
    if (i < n) p[i] = val;
}

// ---- fused QKVS GEMM: [20000,512] x [512,2048] via v_wmma_f32_16x16x32_bf16 ----
__global__ void bgp_qkvs_gemm(const float* __restrict__ nodes, const __bf16* __restrict__ Wp,
                              const float* __restrict__ bq, const float* __restrict__ bk,
                              const float* __restrict__ bv, const float* __restrict__ bs,
                              float* __restrict__ q, float* __restrict__ k,
                              float* __restrict__ v, float* __restrict__ s) {
    const int lane = threadIdx.x & 31;
    const int wave = threadIdx.x >> 5;
    const int rowTile  = blockIdx.x;                 // 0..1249
    const int colGroup = blockIdx.y;                 // 0..3 (one weight segment each)
    const int colBase  = colGroup * 512 + wave * 64; // wave: 16 rows x 64 cols
    const int row = rowTile * 16 + (lane & 15);
    const int hi  = lane >> 4;                       // K-half select per A layout
    const int ntile0 = colBase >> 4;

    v8f acc[4] = {v8f{}, v8f{}, v8f{}, v8f{}};

    for (int kt = 0; kt < KT; ++kt) {
        const float* arow = nodes + (size_t)row * DIMM + kt * 32;
        const int kb = hi ? 16 : 0;
        v16bf a;
        #pragma unroll
        for (int j = 0; j < 8; ++j) a[j]     = (__bf16)arow[kb + j];
        #pragma unroll
        for (int j = 0; j < 8; ++j) a[8 + j] = (__bf16)arow[kb + 8 + j];
        #pragma unroll
        for (int t = 0; t < 4; ++t) {
            const v16bf b = *(const v16bf*)(Wp + (((size_t)kt * NT + (ntile0 + t)) * 32 + lane) * 16);
            acc[t] = __builtin_amdgcn_wmma_f32_16x16x32_bf16(
                false, a, false, b, (short)0, acc[t], false, false);
        }
    }

    const float* bias = (colGroup == 0) ? bq : (colGroup == 1) ? bk : (colGroup == 2) ? bv : bs;
    float* dst = (colGroup == 0) ? q : (colGroup == 1) ? k : (colGroup == 2) ? v : s;
    #pragma unroll
    for (int t = 0; t < 4; ++t) {
        int cid = (colBase & 511) + t * 16 + (lane & 15);
        float bb = bias[cid];
        #pragma unroll
        for (int j = 0; j < 8; ++j) {
            int m = rowTile * 16 + j + (hi ? 8 : 0);
            dst[(size_t)m * DIMM + cid] = acc[t][j] + bb;
        }
    }
}

// ---- per-(edge,head) attention score ----
__global__ void bgp_scores(const float* __restrict__ q, const float* __restrict__ k,
                           const int* __restrict__ src, const int* __restrict__ dst,
                           float* __restrict__ sc) {
    int idx = blockIdx.x * blockDim.x + threadIdx.x;
    if (idx >= EE * HH) return;
    int e = idx >> 3, h = idx & 7;
    const float4* qp = (const float4*)(q + (size_t)dst[e] * DIMM + h * 64);
    const float4* kp = (const float4*)(k + (size_t)src[e] * DIMM + h * 64);
    float acc = 0.f;
    #pragma unroll
    for (int j = 0; j < 16; ++j) {
        float4 a = qp[j], b = kp[j];
        acc += a.x * b.x + a.y * b.y + a.z * b.z + a.w * b.w;
    }
    sc[idx] = acc * 0.125f;   // 1/sqrt(64)
}

__global__ void bgp_segmax(const float* __restrict__ sc, const int* __restrict__ dst,
                           unsigned* __restrict__ mx) {
    int idx = blockIdx.x * blockDim.x + threadIdx.x;
    if (idx >= EE * HH) return;
    int e = idx >> 3, h = idx & 7;
    atomicMax(&mx[(size_t)dst[e] * HH + h], encOrd(sc[idx]));
}

__global__ void bgp_exp(float* __restrict__ sc, const int* __restrict__ dst,
                        const unsigned* __restrict__ mx, float* __restrict__ den) {
    int idx = blockIdx.x * blockDim.x + threadIdx.x;
    if (idx >= EE * HH) return;
    int e = idx >> 3, h = idx & 7;
    float m  = decOrd(mx[(size_t)dst[e] * HH + h]);
    float ex = __expf(sc[idx] - m);
    sc[idx] = ex;
    atomAddF(&den[(size_t)dst[e] * HH + h], ex);
}

__global__ void bgp_agg(const float* __restrict__ ex, const int* __restrict__ src,
                        const int* __restrict__ dst, const float* __restrict__ den,
                        const float* __restrict__ v, float* __restrict__ agg) {
    int idx = blockIdx.x * blockDim.x + threadIdx.x;
    if (idx >= EE * HH) return;
    int e = idx >> 3, h = idx & 7;
    float alpha = ex[idx] / (den[(size_t)dst[e] * HH + h] + 1e-16f);
    const float* vp = v   + (size_t)src[e] * DIMM + h * 64;
    float*       ap = agg + (size_t)dst[e] * DIMM + h * 64;
    #pragma unroll 8
    for (int c = 0; c < 64; ++c) atomAddF(&ap[c], alpha * vp[c]);
}

__global__ void bgp_final(const float* __restrict__ agg, const float* __restrict__ skip,
                          const float* __restrict__ Wp, const float* __restrict__ bp,
                          float* __restrict__ out) {
    int n = blockIdx.x * blockDim.x + threadIdx.x;
    if (n >= NN) return;
    float a0 = bp[0], a1 = bp[1];
    const float* ap = agg  + (size_t)n * DIMM;
    const float* sp = skip + (size_t)n * DIMM;
    for (int d = 0; d < DIMM; ++d) {
        float o = ap[d] + sp[d];
        a0 += o * Wp[2 * d];
        a1 += o * Wp[2 * d + 1];
    }
    out[n] = a0;
    out[NN + n] = a1;
}

extern "C" void kernel_launch(void* const* d_in, const int* in_sizes, int n_in,
                              void* d_out, int out_size, void* d_ws, size_t ws_size,
                              hipStream_t stream) {
    const float* nodes = (const float*)d_in[0];
    const int*   eidx  = (const int*)d_in[1];
    const int*   src   = eidx;            // edge_index[0]
    const int*   dst   = eidx + EE;       // edge_index[1]
    const float* Wq = (const float*)d_in[2];  const float* bq = (const float*)d_in[3];
    const float* Wk = (const float*)d_in[4];  const float* bk = (const float*)d_in[5];
    const float* Wv = (const float*)d_in[6];  const float* bv = (const float*)d_in[7];
    const float* Ws = (const float*)d_in[8];  const float* bs = (const float*)d_in[9];
    const float* Wproj = (const float*)d_in[10];
    const float* bproj = (const float*)d_in[11];
    float* out = (float*)d_out;

    float* ws = (float*)d_ws;
    float*    q   = ws + OFF_Q;
    float*    k   = ws + OFF_K;
    float*    v   = ws + OFF_V;
    float*    sk  = ws + OFF_S;
    float*    agg = ws + OFF_AGG;
    float*    sc  = ws + OFF_SC;
    unsigned* mx  = (unsigned*)(ws + OFF_MAX);
    float*    den = ws + OFF_DEN;
    __bf16*   Wp  = (__bf16*)(ws + OFF_WP);

    // 1. pack weights to bf16 B-fragments
    bgp_pack_w<<<(KT * NT * 32 + 255) / 256, 256, 0, stream>>>(Wq, Wk, Wv, Ws, Wp);

    // 2. zero agg, and segmax(enc=0)+denom (contiguous)
    {
        size_t na = (size_t)NN * DIMM;
        bgp_fill<<<(int)((na + 255) / 256), 256, 0, stream>>>(agg, 0.f, na);
        size_t nm = (size_t)NN * HH * 2;   // mx then den, contiguous
        bgp_fill<<<(int)((nm + 255) / 256), 256, 0, stream>>>(ws + OFF_MAX, 0.f, nm);
    }

    // 3. fused QKV+skip GEMM (WMMA bf16)
    bgp_qkvs_gemm<<<dim3(NN / 16, 4), 256, 0, stream>>>(nodes, Wp, bq, bk, bv, bs, q, k, v, sk);

    // 4-7. edge attention
    int eb = (EE * HH + 255) / 256;
    bgp_scores<<<eb, 256, 0, stream>>>(q, k, src, dst, sc);
    bgp_segmax<<<eb, 256, 0, stream>>>(sc, dst, mx);
    bgp_exp   <<<eb, 256, 0, stream>>>(sc, dst, mx, den);
    bgp_agg   <<<eb, 256, 0, stream>>>(sc, src, dst, den, v, agg);

    // 8. skip-add + 2-col projection
    bgp_final<<<(NN + 255) / 256, 256, 0, stream>>>(agg, sk, Wproj, bproj, out);
}
